// WillingnessWeight_50620484551277
// MI455X (gfx1250) — matrix operand order repaired and checked by reading, not compiled, not run on hardware
//
#include <hip/hip_runtime.h>
#include <hip/hip_bf16.h>
#include <stdint.h>

// ---------------- CDNA5 WMMA plumbing ----------------
typedef __attribute__((ext_vector_type(16))) __bf16 v16bf;
typedef __attribute__((ext_vector_type(8)))  float  v8f;

struct __align__(16) U4 { uint32_t x, y, z, w; };

union BFfrag {
    v16bf    v;
    uint32_t u[8];
    U4       q[2];
};

__device__ __forceinline__ uint32_t pack_bf16(float a, float b) {
    __hip_bfloat162 h = __float22bfloat162_rn(float2{a, b});
    uint32_t r;
    __builtin_memcpy(&r, &h, 4);
    return r;
}
__device__ __forceinline__ float bf2f(uint32_t h) {
    return __uint_as_float((h & 0xFFFFu) << 16);
}

__device__ __forceinline__ v8f wmma_bf16(const BFfrag& a, const BFfrag& b, v8f c) {
    return __builtin_amdgcn_wmma_f32_16x16x32_bf16(
        false, a.v, false, b.v, (short)0, c, false, false);
}

#define LOG2E 1.4426950408889634f

// Combine two softmax-entropy states in base-2 units: (m, Z, T)
// Z = sum 2^(a-m), T = sum 2^(a-m)*(a-m)
__device__ __forceinline__ void combine_state(float& m1, float& z1, float& t1,
                                              float m2, float z2, float t2) {
    float nm = fmaxf(m1, m2);
    float d1 = fmaxf(m1 - nm, -100.f);
    float d2 = fmaxf(m2 - nm, -100.f);
    float e1 = exp2f(d1), e2 = exp2f(d2);
    t1 = e1 * (t1 + d1 * z1) + e2 * (t2 + d2 * z2);
    z1 = e1 * z1 + e2 * z2;
    m1 = nm;
}

// =====================================================================
// Projection: O[M x 128] (bf16, row-major) = X @ W^T + bias, WMMA bf16.
// A = W tile (LDS), B = X^T (registers): C fragment holds 8 consecutive
// dout per lane -> one b128 store per 16x16 tile.
// =====================================================================
__global__ __launch_bounds__(256) void proj_kernel(
    const float* __restrict__ X, const float* __restrict__ W,
    const float* __restrict__ bias, uint32_t* __restrict__ Ow, int M)
{
    __shared__ __align__(16) uint32_t ldsW[128 * 64]; // [dout][din/2] bf16 pairs

    const int tid = threadIdx.x;
    for (int i = tid; i < 128 * 64; i += 256)
        ldsW[i] = pack_bf16(W[2 * i], W[2 * i + 1]);
    __syncthreads();

    const int wave = tid >> 5, lane = tid & 31;
    const int n = lane & 15, hi = lane >> 4;

    for (int s = 0; s < 4; ++s) {
        int strip = (blockIdx.x * 8 + wave) * 4 + s;
        if (strip * 16 >= M) break;                    // wave-uniform

        const float* xr = X + (size_t)(strip * 16 + n) * 128;
        BFfrag bfrag[4];
        #pragma unroll
        for (int ks = 0; ks < 4; ++ks) {
            int dbase = ks * 32 + hi * 16;
            #pragma unroll
            for (int v = 0; v < 8; ++v) {
                float2 xv = *(const float2*)(xr + dbase + 2 * v);
                bfrag[ks].u[v] = pack_bf16(xv.x, xv.y);
            }
        }

        #pragma unroll
        for (int dt = 0; dt < 8; ++dt) {
            v8f c;
            if (bias) {
                float4 b0 = *(const float4*)(bias + dt * 16 + hi * 8);
                float4 b1 = *(const float4*)(bias + dt * 16 + hi * 8 + 4);
                c[0] = b0.x; c[1] = b0.y; c[2] = b0.z; c[3] = b0.w;
                c[4] = b1.x; c[5] = b1.y; c[6] = b1.z; c[7] = b1.w;
            } else {
                #pragma unroll
                for (int v = 0; v < 8; ++v) c[v] = 0.f;
            }
            #pragma unroll
            for (int ks = 0; ks < 4; ++ks) {
                BFfrag a;                               // A: W rows dt*16..+15
                int base = (dt * 16 + n) * 64 + ks * 16 + hi * 8;
                a.q[0] = *(const U4*)(&ldsW[base]);
                a.q[1] = *(const U4*)(&ldsW[base + 4]);
                c = wmma_bf16(a, bfrag[ks], c);
            }
            U4 o;
            o.x = pack_bf16(c[0], c[1]);
            o.y = pack_bf16(c[2], c[3]);
            o.z = pack_bf16(c[4], c[5]);
            o.w = pack_bf16(c[6], c[7]);
            *(U4*)(Ow + (size_t)(strip * 16 + n) * 64 + dt * 8 + hi * 4) = o;
        }
    }
}

// =====================================================================
// alpha_self[row] = <Q,Kc>, qb[row] = <Q, Wk_b>; wave per 4 rows, coalesced.
// =====================================================================
__global__ __launch_bounds__(256) void alpha_kernel(
    const uint32_t* __restrict__ Qw, const uint32_t* __restrict__ Kw,
    const float* __restrict__ Wkb, float* __restrict__ alpha_self,
    float* __restrict__ qb)
{
    const int wave = threadIdx.x >> 5, lane = threadIdx.x & 31;
    float4 wb = *(const float4*)(Wkb + 4 * lane);
    for (int r = 0; r < 4; ++r) {
        int row = (blockIdx.x * 8 + wave) * 4 + r;
        uint2 qw = *(const uint2*)(Qw + (size_t)row * 64 + 2 * lane);
        uint2 kw = *(const uint2*)(Kw + (size_t)row * 64 + 2 * lane);
        float q0 = bf2f(qw.x), q1 = bf2f(qw.x >> 16);
        float q2 = bf2f(qw.y), q3 = bf2f(qw.y >> 16);
        float sa = q0 * bf2f(kw.x) + q1 * bf2f(kw.x >> 16)
                 + q2 * bf2f(kw.y) + q3 * bf2f(kw.y >> 16);
        float sq = q0 * wb.x + q1 * wb.y + q2 * wb.z + q3 * wb.w;
        #pragma unroll
        for (int off = 16; off >= 1; off >>= 1) {
            sa += __shfl_xor(sa, off, 32);
            sq += __shfl_xor(sq, off, 32);
        }
        if (lane == 0) { alpha_self[row] = sa; qb[row] = sq; }
    }
}

// =====================================================================
// Core: S = Q @ Ku^T via WMMA (A = Ku tile in LDS, B = Q^T resident),
// fused mask*S softmax partials with fixed base (qb cancels out!).
// NU split in 2 halves for 2x wave parallelism; partial (Z,T) just add.
// Grid: 128 b x 2 jt x 2 half = 512 blocks x 256 thr.
// =====================================================================
__global__ __launch_bounds__(256) void score_kernel(
    const uint32_t* __restrict__ Qw, const uint32_t* __restrict__ Kubf,
    const int* __restrict__ mask, float* __restrict__ zpart,
    float* __restrict__ tpart)
{
    __shared__ __align__(16) uint32_t ldsK[64 * 64];   // 64 k' x 128 bf16, 16KB

    const int tid = threadIdx.x, wave = tid >> 5, lane = tid & 31;
    const int n = lane & 15, hi = lane >> 4;
    const int b = blockIdx.x >> 2;
    const int jBase = ((blockIdx.x >> 1) & 1) * 128 + wave * 16;
    const int half = blockIdx.x & 1;
    const int j = jBase + n;
    const int ktBeg = half * 1024, ktEnd = ktBeg + 1024;

    // Resident B fragments: Q row j (packed K-pairs along d)
    const uint32_t* qrow = Qw + (size_t)(b * 256 + j) * 64;
    BFfrag bfrag[4];
    #pragma unroll
    for (int ks = 0; ks < 4; ++ks) {
        int base = ks * 16 + hi * 8;
        bfrag[ks].q[0] = *(const U4*)(qrow + base);
        bfrag[ks].q[1] = *(const U4*)(qrow + base + 4);
    }

    // fixed softmax base m = qb*log2e + 32  =>  d = mask*S*log2e - 32
    float z = 0.f, t = 0.f;

    const int* mlane = mask + (size_t)(b * 256 + j) * 2048;
    const uint32_t* Kw = Kubf + (size_t)b * 2048 * 64;
    const uint32_t lds0 = (uint32_t)(uintptr_t)(&ldsK[0]);

    for (int kt = ktBeg; kt < ktEnd; kt += 64) {
        // Async-copy the 16 KB Ku tile into LDS (CDNA5 ASYNCcnt path)
        {
            const uint32_t* src = Kw + (size_t)kt * 64;
            #pragma unroll
            for (int i = 0; i < 4; ++i) {
                int woff = (tid + 256 * i) * 4;               // word offset
                uint64_t ga = (uint64_t)(uintptr_t)(src + woff);
                uint32_t la = lds0 + (uint32_t)woff * 4u;     // LDS byte addr
                asm volatile("global_load_async_to_lds_b128 %0, %1, off"
                             :: "v"(la), "v"(ga) : "memory");
            }
        }
        if (kt + 64 < ktEnd)
            __builtin_prefetch(Kw + (size_t)(kt + 64) * 64 + tid * 16, 0, 0);
        asm volatile("s_wait_asynccnt 0x0" ::: "memory");
        __syncthreads();

        #pragma unroll
        for (int nt = 0; nt < 4; ++nt) {
            v8f c;
            #pragma unroll
            for (int v = 0; v < 8; ++v) c[v] = 0.f;
            #pragma unroll
            for (int ks = 0; ks < 4; ++ks) {
                BFfrag a;                              // A: Ku rows nt*16..+15
                int base = (nt * 16 + n) * 64 + ks * 16 + hi * 8;
                a.q[0] = *(const U4*)(&ldsK[base]);
                a.q[1] = *(const U4*)(&ldsK[base + 4]);
                c = wmma_bf16(a, bfrag[ks], c);
            }
            // lane's k' columns: kt + nt*16 + 8*hi + {0..7}  (contiguous)
            int4 m0 = *(const int4*)(mlane + kt + nt * 16 + 8 * hi);
            int4 m1 = *(const int4*)(mlane + kt + nt * 16 + 8 * hi + 4);
            int mk[8] = { m0.x, m0.y, m0.z, m0.w, m1.x, m1.y, m1.z, m1.w };
            #pragma unroll
            for (int v = 0; v < 8; ++v) {
                float d = fmaf((float)mk[v] * c[v], LOG2E, -32.0f);
                float e = exp2f(d);
                z += e;
                t = fmaf(e, d, t);
            }
        }
        __syncthreads();
    }

    // lane pair (hi=0 / hi=1) shares row j and identical base -> plain add
    z += __shfl_xor(z, 16, 32);
    t += __shfl_xor(t, 16, 32);
    if (hi == 0) {
        zpart[half * 32768 + b * 256 + j] = z;
        tpart[half * 32768 + b * 256 + j] = t;
    }
}

// =====================================================================
// Final: combine NU-half partials, fold in alpha_self, entropy, then
// normalize over NC per batch. One block per batch row.
// =====================================================================
__global__ __launch_bounds__(256) void final_kernel(
    const float* __restrict__ zpart, const float* __restrict__ tpart,
    const float* __restrict__ alpha_self, const float* __restrict__ qb,
    float* __restrict__ out)
{
    __shared__ float red[256];
    const int b = blockIdx.x, jj = threadIdx.x;
    const int idx = b * 256 + jj;

    float z = zpart[idx] + zpart[32768 + idx];
    float t = tpart[idx] + tpart[32768 + idx];
    float mm = qb[idx] * LOG2E + 32.0f;                // the fixed base
    combine_state(mm, z, t, alpha_self[idx] * LOG2E, 1.0f, 0.0f);
    float H = __log2f(z) - t / z;                      // entropy in bits

    red[jj] = H;
    __syncthreads();
    for (int s = 128; s > 0; s >>= 1) {
        if (jj < s) red[jj] += red[jj + s];
        __syncthreads();
    }
    out[idx] = H / red[0];
}

// =====================================================================
extern "C" void kernel_launch(void* const* d_in, const int* in_sizes, int n_in,
                              void* d_out, int out_size, void* d_ws, size_t ws_size,
                              hipStream_t stream)
{
    (void)in_sizes; (void)n_in; (void)out_size; (void)ws_size;
    const float* clicked   = (const float*)d_in[0];
    const float* unclicked = (const float*)d_in[1];
    const int*   mask      = (const int*)d_in[2];
    const float* Wq_w      = (const float*)d_in[3];
    const float* Wq_b      = (const float*)d_in[4];
    const float* Wk_w      = (const float*)d_in[5];
    const float* Wk_b      = (const float*)d_in[6];
    float* out = (float*)d_out;

    char* ws = (char*)d_ws;
    uint32_t* Qbf   = (uint32_t*)(ws);                              //  8 MiB
    uint32_t* Kcbf  = (uint32_t*)(ws + ((size_t)8  << 20));         //  8 MiB
    uint32_t* Kubf  = (uint32_t*)(ws + ((size_t)16 << 20));         // 64 MiB
    char*     tail  = ws + ((size_t)80 << 20);
    float*    alpha = (float*)(tail);                               // 128 KiB
    float*    qb    = (float*)(tail + (128u << 10));                // 128 KiB
    float*    zpart = (float*)(tail + (256u << 10));                // 256 KiB
    float*    tpart = (float*)(tail + (512u << 10));                // 256 KiB

    proj_kernel<<<64,  256, 0, stream>>>(clicked,   Wq_w, Wq_b,   Qbf,  128 * 256);
    proj_kernel<<<64,  256, 0, stream>>>(clicked,   Wk_w, Wk_b,   Kcbf, 128 * 256);
    proj_kernel<<<512, 256, 0, stream>>>(unclicked, Wk_w, nullptr, Kubf, 128 * 2048);

    alpha_kernel<<<1024, 256, 0, stream>>>(Qbf, Kcbf, Wk_b, alpha, qb);

    score_kernel<<<512, 256, 0, stream>>>(Qbf, Kubf, mask, zpart, tpart);

    final_kernel<<<128, 256, 0, stream>>>(zpart, tpart, alpha, qb, out);
}
